// Online_SRNN_GLN_Cell_14551349199507
// MI455X (gfx1250) — matrix operand order repaired and checked by reading, not compiled, not run on hardware
//
#include <hip/hip_runtime.h>

typedef __attribute__((ext_vector_type(16))) _Float16 v16h;
typedef __attribute__((ext_vector_type(8)))  _Float16 v8h;
typedef __attribute__((ext_vector_type(4)))  _Float16 v4h;
typedef __attribute__((ext_vector_type(8)))  float    v8f;

#define SEQ   8192
#define INSZ  512
#define HYPER 1024
#define HID   4096

__device__ __forceinline__ float sigmoidf(float x) {
    return 1.0f / (1.0f + __expf(-x));
}

// ---------------------------------------------------------------- convert
// n must be a multiple of 4 (true for every tensor here).
__global__ void f32_to_f16_v4(const float4* __restrict__ src,
                              v4h* __restrict__ dst, int n4) {
    int i = blockIdx.x * blockDim.x + threadIdx.x;
    if (i < n4) {
        float4 f = src[i];
        v4h h;
        h[0] = (_Float16)f.x; h[1] = (_Float16)f.y;
        h[2] = (_Float16)f.z; h[3] = (_Float16)f.w;
        dst[i] = h;
    }
}

// ---------------------------------------------------------------- WMMA GEMM
// C[m,n] = sum_k A[m,k] * W[n,k]  (i.e. A @ W^T), f16 in, f32 accumulate.
// MODE 0: out_h = f16(v + bv)
// MODE 1: out_h = f16((v+bv) * sigmoid(g+bg))          (two B matrices)
// MODE 2: out_h = f16((v+bv) * sigmoid(v+bv))          (shared-weight pair)
// MODE 3: out_h = f16(sigmoid(v+bv))                   (multihead gate)
// MODE 4: out_f = (v+bv) * float(mul[m,n])             (pre-scan b, f32 out)
template <int MODE>
__global__ __launch_bounds__(256)
void gemm_wmma(const _Float16* __restrict__ A,
               const _Float16* __restrict__ Wv,
               const float*    __restrict__ bv,
               const _Float16* __restrict__ Wg,
               const float*    __restrict__ bg,
               const _Float16* __restrict__ mul,
               _Float16*       __restrict__ outH,
               float*          __restrict__ outF,
               int M, int N, int K)
{
    const int tid   = threadIdx.x;
    const int lane  = tid & 31;
    const int wave  = tid >> 5;
    const int waveM = wave & 3;   // 4 waves along M
    const int waveN = wave >> 2;  // 2 waves along N
    const int bm = blockIdx.y * 128 + waveM * 32;   // wave: 32 rows
    const int bn = blockIdx.x * 128 + waveN * 64;   // wave: 64 cols

    // A fragment addressing (16-bit A 16x32, contiguous k order 16*(lane>>4) + e)
    const int arow = lane & 15;
    const int akb  = (lane >> 4) << 4;   // 0 or 16
    // B fragment addressing (16-bit B 32x16, contiguous k order 16*(lane>>4) + e), B[k][n] = W[n][k]
    const int bcol = lane & 15;
    const int bkb  = (lane >> 4) << 4;   // 0 or 16

    v8f accV[2][4];
    v8f accG[2][4];
    const v8f vzero = {0.f,0.f,0.f,0.f,0.f,0.f,0.f,0.f};
#pragma unroll
    for (int i = 0; i < 2; ++i)
#pragma unroll
        for (int j = 0; j < 4; ++j) { accV[i][j] = vzero; accG[i][j] = vzero; }

    for (int k0 = 0; k0 < K; k0 += 32) {
        v16h afrag[2];
#pragma unroll
        for (int mi = 0; mi < 2; ++mi) {
            const _Float16* pa = A + (size_t)(bm + mi * 16 + arow) * K + k0 + akb;
            v8h lo = *(const v8h*)pa;          // K = kb .. kb+7
            v8h hi = *(const v8h*)(pa + 8);    // K = kb+8 .. kb+15
            v16h a;
#pragma unroll
            for (int e = 0; e < 8; ++e) { a[e] = lo[e]; a[e + 8] = hi[e]; }
            afrag[mi] = a;
        }
#pragma unroll
        for (int nf = 0; nf < 4; ++nf) {
            const _Float16* pb = Wv + (size_t)(bn + nf * 16 + bcol) * K + k0 + bkb;
            v16h b = *(const v16h*)pb;         // 16 contiguous K values
#pragma unroll
            for (int mi = 0; mi < 2; ++mi)
                accV[mi][nf] = __builtin_amdgcn_wmma_f32_16x16x32_f16(
                    false, afrag[mi], false, b, (short)0, accV[mi][nf], false, false);
            if (MODE == 1) {
                const _Float16* pg = Wg + (size_t)(bn + nf * 16 + bcol) * K + k0 + bkb;
                v16h g = *(const v16h*)pg;
#pragma unroll
                for (int mi = 0; mi < 2; ++mi)
                    accG[mi][nf] = __builtin_amdgcn_wmma_f32_16x16x32_f16(
                        false, afrag[mi], false, g, (short)0, accG[mi][nf], false, false);
            }
        }
    }

    // C/D layout: row = v + 8*(lane>>4), col = lane&15
    const int rofs = (lane >> 4) * 8;
    const int ccol = lane & 15;
#pragma unroll
    for (int mi = 0; mi < 2; ++mi) {
#pragma unroll
        for (int nf = 0; nf < 4; ++nf) {
            const int ncol  = bn + nf * 16 + ccol;
            const float bvv = bv[ncol];
            const float bgg = (MODE == 1) ? bg[ncol] : 0.0f;
#pragma unroll
            for (int e = 0; e < 8; ++e) {
                const int    mrow = bm + mi * 16 + rofs + e;
                const size_t idx  = (size_t)mrow * N + ncol;
                const float  v    = accV[mi][nf][e] + bvv;
                if (MODE == 0) {
                    outH[idx] = (_Float16)v;
                } else if (MODE == 1) {
                    const float g = accG[mi][nf][e] + bgg;
                    outH[idx] = (_Float16)(v * sigmoidf(g));
                } else if (MODE == 2) {
                    outH[idx] = (_Float16)(v * sigmoidf(v));
                } else if (MODE == 3) {
                    outH[idx] = (_Float16)sigmoidf(v);
                } else {
                    outF[idx] = v * (float)mul[idx];
                }
            }
        }
    }
}

// ---------------------------------------------------------------- diagonal scan
// h_t[j] = relu(b_t[j] + h_{t-1}[(j-1) mod H]).
// Dependency structure is diagonal: h_t[j] depends only on h_{t-1}[j-1], so the
// computation splits into HID=4096 fully independent chains. Chain c visits
// j = (c+t) mod H at step t, seeded with hidden[(c-1) mod H]. One thread per
// chain: zero barriers, loads/stores stay 128B-coalesced per wave (row t is
// just rotated by t), and addresses are loop-independent so loads pipeline.
// In place over d_out rows; h_last lands at d_out + SEQ*HID.
__global__ __launch_bounds__(128)
void scan_diag(float* __restrict__ out, const float* __restrict__ h0)
{
    const int c = blockIdx.x * blockDim.x + threadIdx.x;   // 0..HID-1
    float v = h0[(c + HID - 1) & (HID - 1)];               // h_{-1}[c-1]
    int j = c;
#pragma unroll 4
    for (int t = 0; t < SEQ; ++t) {
        float* p = out + (size_t)t * HID + j;
        v = fmaxf(*p + v, 0.0f);
        *p = v;
        j = (j + 1) & (HID - 1);
    }
    out[(size_t)SEQ * HID + ((c + SEQ - 1) & (HID - 1))] = v;  // h_last
}

// ---------------------------------------------------------------- launch
static inline void cvt(const float* s, _Float16* d, int n, hipStream_t st) {
    int n4 = n / 4;
    f32_to_f16_v4<<<(n4 + 255) / 256, 256, 0, st>>>((const float4*)s, (v4h*)d, n4);
}

extern "C" void kernel_launch(void* const* d_in, const int* in_sizes, int n_in,
                              void* d_out, int out_size, void* d_ws, size_t ws_size,
                              hipStream_t stream) {
    const float* x      = (const float*)d_in[0];
    const float* hidden = (const float*)d_in[1];
    const float* w_in   = (const float*)d_in[2];
    const float* b_in   = (const float*)d_in[3];
    const float* wg0    = (const float*)d_in[4];
    const float* bg0    = (const float*)d_in[5];
    const float* wg0g   = (const float*)d_in[6];
    const float* bg0g   = (const float*)d_in[7];
    const float* wg_sh  = (const float*)d_in[8];
    const float* bg_sh  = (const float*)d_in[9];
    const float* w_h    = (const float*)d_in[10];
    const float* b_h    = (const float*)d_in[11];
    const float* w2     = (const float*)d_in[12];
    const float* b2     = (const float*)d_in[13];

    char* ws = (char*)d_ws;
    _Float16* xh    = (_Float16*)ws; ws += (size_t)SEQ * INSZ * 2;
    _Float16* w_inH = (_Float16*)ws; ws += (size_t)HYPER * INSZ * 2;
    _Float16* wg0H  = (_Float16*)ws; ws += (size_t)HYPER * HYPER * 2;
    _Float16* wg0gH = (_Float16*)ws; ws += (size_t)HYPER * HYPER * 2;
    _Float16* wgshH = (_Float16*)ws; ws += (size_t)HYPER * HYPER * 2;
    _Float16* w_hH  = (_Float16*)ws; ws += (size_t)HID * HYPER * 2;
    _Float16* w2H   = (_Float16*)ws; ws += (size_t)HID * INSZ * 2;
    _Float16* h1    = (_Float16*)ws; ws += (size_t)SEQ * HYPER * 2;
    _Float16* h2    = (_Float16*)ws; ws += (size_t)SEQ * HYPER * 2;
    _Float16* h3    = (_Float16*)ws; ws += (size_t)SEQ * HYPER * 2;
    _Float16* sig   = (_Float16*)ws; ws += (size_t)SEQ * HID * 2;

    // f32 -> f16 staging
    cvt(x,     xh,    SEQ * INSZ,     stream);
    cvt(w_in,  w_inH, HYPER * INSZ,   stream);
    cvt(wg0,   wg0H,  HYPER * HYPER,  stream);
    cvt(wg0g,  wg0gH, HYPER * HYPER,  stream);
    cvt(wg_sh, wgshH, HYPER * HYPER,  stream);
    cvt(w_h,   w_hH,  HID * HYPER,    stream);
    cvt(w2,    w2H,   HID * INSZ,     stream);

    float* outF = (float*)d_out;
    dim3 blk(256);

    // h1 = f16(x @ w_in^T + b_in)
    gemm_wmma<0><<<dim3(HYPER / 128, SEQ / 128), blk, 0, stream>>>(
        xh, w_inH, b_in, nullptr, nullptr, nullptr, h1, nullptr, SEQ, HYPER, INSZ);
    // h2 = f16((h1@wg0^T+bg0) * sigmoid(h1@wg0g^T+bg0g))
    gemm_wmma<1><<<dim3(HYPER / 128, SEQ / 128), blk, 0, stream>>>(
        h1, wg0H, bg0, wg0gH, bg0g, nullptr, h2, nullptr, SEQ, HYPER, HYPER);
    // h3 = f16(v * sigmoid(v)), v = h2@wg_sh^T+bg_sh   (shared-weight pair)
    gemm_wmma<2><<<dim3(HYPER / 128, SEQ / 128), blk, 0, stream>>>(
        h2, wgshH, bg_sh, nullptr, nullptr, nullptr, h3, nullptr, SEQ, HYPER, HYPER);
    // sig = f16(sigmoid(x @ w2^T + b2))   (multihead gate)
    gemm_wmma<3><<<dim3(HID / 128, SEQ / 128), blk, 0, stream>>>(
        xh, w2H, b2, nullptr, nullptr, nullptr, sig, nullptr, SEQ, HID, INSZ);
    // b = (h3 @ w_h^T + b_h) * sig  -> f32, written straight into d_out rows
    gemm_wmma<4><<<dim3(HID / 128, SEQ / 128), blk, 0, stream>>>(
        h3, w_hH, b_h, nullptr, nullptr, sig, nullptr, outF, SEQ, HID, HYPER);

    // diagonal-parallel relu scan: 4096 independent chains, one thread each
    scan_diag<<<dim3(HID / 128), dim3(128), 0, stream>>>(outF, hidden);
}
